// GRU_60644938219590
// MI455X (gfx1250) — matrix operand order repaired and checked by reading, not compiled, not run on hardware
//
#include <hip/hip_runtime.h>
#include <hip/hip_bf16.h>
#include <math.h>

// ---------------------------------------------------------------------------
// Problem constants (from reference): B=32, T=12, N=2048, DIN=1, H=64, DE=10, K=3
// Unified padded channel width CPAD=128 for both layers (layer0 uses cols 0..64).
// Activations stored [N, B, CPAD] bf16  ==  [2048 x 4096] node-matrix view
//                                      ==  [65536 x 128] row-matrix view.
// ---------------------------------------------------------------------------
#define NN    2048
#define BB    32
#define TT    12
#define HH    64
#define DEE   10
#define CPAD  128
#define RR    (NN * BB)          // 65536 rows (n-major, b-minor)
#define WCOLS (BB * CPAD)        // 4096
#define KTOT  (DEE * 3 * CPAD)   // 3840 : pool-GEMM reduction dim (d,k-hop,channel)

// gfx1250 async global->LDS path (ASYNCcnt), guarded so absence falls back to
// the proven global->VGPR->ds_store path.
#if defined(__has_builtin)
#if __has_builtin(__builtin_amdgcn_global_load_async_to_lds_b128) && \
    __has_builtin(__builtin_amdgcn_s_wait_asynccnt)
#define USE_ASYNC_LDS 1
#endif
#endif
#ifndef USE_ASYNC_LDS
#define USE_ASYNC_LDS 0
#endif

#define AS1 __attribute__((address_space(1)))
#define AS3 __attribute__((address_space(3)))

// Matches the builtin's parameter type: __attribute__((__vector_size__(16))) int *
typedef int v4i_vec __attribute__((__vector_size__(16)));

typedef __attribute__((ext_vector_type(16))) __bf16 v16bf;
typedef __attribute__((ext_vector_type(8)))  float  v8f;

union Frag32B { uint4 u[2]; v16bf v; };

// WMMA 16x16x32 bf16 A-fragment from a row-major LDS tile (32 bf16 = 64B/row).
// lanes0-15 row M=lane, K0..15; lanes16-31 row M=lane-16, K16..31 (contiguous 32B)
static __device__ inline v16bf load_a_frag(const __bf16* base, int lane) {
  const char* p = (const char*)base + (lane & 15) * 64 + ((lane >> 4) & 1) * 32;
  Frag32B f;
  f.u[0] = *(const uint4*)(p);
  f.u[1] = *(const uint4*)(p + 16);
  return f.v;
}
// B-fragment from a col-major LDS tile (32 bf16 = 64B/col).
// lanes0-15 col N=lane, K0..15; lanes16-31 col N=lane-16, K16..31 (contiguous 32B)
static __device__ inline v16bf load_b_frag(const __bf16* base, int lane) {
  const char* p = (const char*)base + (lane & 15) * 64 + ((lane >> 4) & 1) * 32;
  Frag32B f;
  f.u[0] = *(const uint4*)(p);
  f.u[1] = *(const uint4*)(p + 16);
  return f.v;
}

static __device__ inline v8f vzero8() {
  v8f z = {0.f, 0.f, 0.f, 0.f, 0.f, 0.f, 0.f, 0.f};
  return z;
}

// ---------------------------------------------------------------------------
// K1: A = softmax(relu(E E^T)) row-wise, written bf16. One block per row.
// ---------------------------------------------------------------------------
__global__ __launch_bounds__(256) void adj_softmax_kernel(const float* __restrict__ E,
                                                          __bf16* __restrict__ A) {
  const int n = blockIdx.x;
  const int tid = threadIdx.x;
  const int lane = tid & 31;
  const int wave = tid >> 5;
  float en[DEE];
#pragma unroll
  for (int d = 0; d < DEE; ++d) en[d] = E[n * DEE + d];

  __shared__ float red[8];
  float psum = 0.f;
  for (int m = tid; m < NN; m += 256) {
    float s = 0.f;
#pragma unroll
    for (int d = 0; d < DEE; ++d) s += en[d] * E[m * DEE + d];
    s = fmaxf(s, 0.f);
    psum += __expf(s);
  }
#pragma unroll
  for (int off = 16; off > 0; off >>= 1) psum += __shfl_down(psum, off, 32);
  if (lane == 0) red[wave] = psum;
  __syncthreads();
  float total = 0.f;
#pragma unroll
  for (int w = 0; w < 8; ++w) total += red[w];
  const float inv = 1.f / total;
  for (int m = tid; m < NN; m += 256) {
    float s = 0.f;
#pragma unroll
    for (int d = 0; d < DEE; ++d) s += en[d] * E[m * DEE + d];
    s = fmaxf(s, 0.f);
    A[(size_t)n * NN + m] = (__bf16)(__expf(s) * inv);
  }
}

// ---------------------------------------------------------------------------
// K2: elementwise f32 -> bf16
// ---------------------------------------------------------------------------
__global__ void cvt_bf16_kernel(const float* __restrict__ src, __bf16* __restrict__ dst, int count) {
  int i = blockIdx.x * blockDim.x + threadIdx.x;
  if (i < count) dst[i] = (__bf16)src[i];
}

// ---------------------------------------------------------------------------
// K3: pool [DE,3,Cin,O] f32 -> padded bf16 [KTOT, O] (channel padded Cin->128)
// ---------------------------------------------------------------------------
__global__ void cvt_pool_kernel(const float* __restrict__ src, __bf16* __restrict__ dst,
                                int Cin, int O) {
  int idx = blockIdx.x * blockDim.x + threadIdx.x;
  if (idx >= KTOT * O) return;
  int k = idx / O, o = idx - k * O;
  int d = k / (3 * CPAD);
  int rem = k - d * (3 * CPAD);
  int kh = rem >> 7;
  int i = rem & (CPAD - 1);
  float v = (i < Cin) ? src[(((size_t)d * 3 + kh) * Cin + i) * O + o] : 0.f;
  dst[idx] = (__bf16)v;
}

// ---------------------------------------------------------------------------
// K4: per-node bias  out[n,o] = sum_d E[n,d] * b[d,o]
// ---------------------------------------------------------------------------
__global__ void bias_kernel(const float* __restrict__ E, const float* __restrict__ b,
                            float* __restrict__ out, int O) {
  int idx = blockIdx.x * blockDim.x + threadIdx.x;
  if (idx >= NN * O) return;
  int n = idx / O, o = idx - n * O;
  float s = 0.f;
#pragma unroll
  for (int d = 0; d < DEE; ++d) s += E[n * DEE + d] * b[d * O + o];
  out[idx] = s;
}

__global__ void zero_f32_kernel(float* __restrict__ p, int count) {
  int i = blockIdx.x * blockDim.x + threadIdx.x;
  if (i < count) p[i] = 0.f;
}

// ---------------------------------------------------------------------------
// K5: build GRU gconv input (bf16, padded).
//   layer0: [x_t | (z*)h0 | 0...]     layer1: [h0 | (z*)h1]
// ---------------------------------------------------------------------------
__global__ void build_in_kernel(const float* __restrict__ x,     // [B,T,N] (DIN=1), layer0 only
                                const float* __restrict__ h_own, // this layer's state [RR,64]
                                const float* __restrict__ h_prev,// layer1: layer0 state [RR,64]
                                const float* __restrict__ zr,    // [RR,128]; z = cols 0..63 (cand only)
                                __bf16* __restrict__ xs,         // [RR,128]
                                int layer, int cand, int t) {
  int idx = blockIdx.x * blockDim.x + threadIdx.x;
  if (idx >= RR * CPAD) return;
  int row = idx >> 7;
  int c = idx & (CPAD - 1);
  float v = 0.f;
  if (layer == 0) {
    if (c == 0) {
      int n = row >> 5, b = row & 31;
      v = x[((size_t)b * TT + t) * NN + n];
    } else if (c < HH + 1) {
      float h = h_own[(size_t)row * HH + (c - 1)];
      if (cand) h *= zr[(size_t)row * CPAD + (c - 1)];
      v = h;
    }
  } else {
    if (c < HH) {
      v = h_prev[(size_t)row * HH + c];
    } else {
      float h = h_own[(size_t)row * HH + (c - HH)];
      if (cand) h *= zr[(size_t)row * CPAD + (c - HH)];
      v = h;
    }
  }
  xs[idx] = (__bf16)v;
}

// ---------------------------------------------------------------------------
// K6: node propagation GEMM  Y[2048,4096] = Adj[2048,2048] @ X[2048,4096] (bf16, f32 acc)
// Macro-tile 128x128, BK=32, 8 waves (2x4), wave = 64x32 (4x2 WMMA tiles).
// Adj tile staged via async global->LDS (no transform); X tile transposed with
// packed ds_store_b32 (two K values per store).
// ---------------------------------------------------------------------------
__global__ __launch_bounds__(256) void prop_gemm_kernel(const __bf16* __restrict__ Adj,
                                                        const __bf16* __restrict__ X,
                                                        __bf16* __restrict__ Y) {
  __shared__ alignas(16) __bf16 lA[128 * 32];   // row-major [row][k]
  __shared__ alignas(16) __bf16 lB[128 * 32];   // col-major [col][k]
  const int tid = threadIdx.x;
  const int lane = tid & 31;
  const int wave = tid >> 5;
  const int wr = wave >> 2;          // 0..1
  const int wc = wave & 3;           // 0..3
  const int rowBase = blockIdx.y * 128;
  const int colBase = blockIdx.x * 128;

  v8f acc[4][2];
#pragma unroll
  for (int i = 0; i < 4; ++i)
#pragma unroll
    for (int j = 0; j < 2; ++j) acc[i][j] = vzero8();

  for (int k0 = 0; k0 < NN; k0 += 32) {
    {   // stage Adj tile 128x32 (each thread: 32 contiguous bytes)
      const int idx = tid * 16;
      const int r = idx >> 5;
      const int kk = idx & 31;
      const __bf16* g = Adj + (size_t)(rowBase + r) * NN + (k0 + kk);
#if USE_ASYNC_LDS
      __builtin_amdgcn_global_load_async_to_lds_b128(
          (AS1 v4i_vec*)(g),
          (AS3 v4i_vec*)(&lA[r * 32 + kk]), 0, 0);
      __builtin_amdgcn_global_load_async_to_lds_b128(
          (AS1 v4i_vec*)(g + 8),
          (AS3 v4i_vec*)(&lA[r * 32 + kk + 8]), 0, 0);
#else
      uint4 v0 = *(const uint4*)(g);
      uint4 v1 = *(const uint4*)(g + 8);
      if (k0 + 32 < NN) __builtin_prefetch(g + 32, 0, 1);   // global_prefetch_b8
      *(uint4*)&lA[r * 32 + kk] = v0;
      *(uint4*)&lA[r * 32 + kk + 8] = v1;
#endif
    }
    {   // stage X tile 32x128, transpose: 2 K-rows packed per ds_store_b32
      const int r0 = (tid >> 4) * 2;      // 0,2,...,30
      const int c0 = (tid & 15) * 8;      // 0..120
      const __bf16* g0 = X + (size_t)(k0 + r0) * WCOLS + colBase + c0;
      alignas(16) unsigned short ta[8], tb[8];
      *(uint4*)ta = *(const uint4*)(g0);
      *(uint4*)tb = *(const uint4*)(g0 + WCOLS);
      if (k0 + 32 < NN) __builtin_prefetch(g0 + (size_t)32 * WCOLS, 0, 1);
#pragma unroll
      for (int i = 0; i < 8; ++i) {
        unsigned pv = (unsigned)ta[i] | ((unsigned)tb[i] << 16);
        *(unsigned*)&lB[(c0 + i) * 32 + r0] = pv;
      }
    }
#if USE_ASYNC_LDS
    __builtin_amdgcn_s_wait_asynccnt(0);
#endif
    __syncthreads();
    v16bf bf[2];
#pragma unroll
    for (int tn = 0; tn < 2; ++tn)
      bf[tn] = load_b_frag(&lB[(wc * 32 + tn * 16) * 32], lane);
#pragma unroll
    for (int tm = 0; tm < 4; ++tm) {
      v16bf a = load_a_frag(&lA[(wr * 64 + tm * 16) * 32], lane);
#pragma unroll
      for (int tn = 0; tn < 2; ++tn)
        acc[tm][tn] = __builtin_amdgcn_wmma_f32_16x16x32_bf16(
            false, a, false, bf[tn], (short)0, acc[tm][tn], false, false);
    }
    __syncthreads();
  }

  const int hi = (lane >> 4) & 1;
  const int cl = lane & 15;
#pragma unroll
  for (int tm = 0; tm < 4; ++tm)
#pragma unroll
    for (int tn = 0; tn < 2; ++tn) {
      const int col = colBase + wc * 32 + tn * 16 + cl;
#pragma unroll
      for (int j = 0; j < 8; ++j) {
        const int row = rowBase + wr * 64 + tm * 16 + j + hi * 8;
        Y[(size_t)row * WCOLS + col] = (__bf16)acc[tm][tn][j];
      }
    }
}

// ---------------------------------------------------------------------------
// K7: fused pool GEMM + E-scaling + bias + activation.
//   out[row,o] = act( sum_{d,kh,i} (E[n,d]*seg_kh[row,i]) * pool[(d,kh,i),o] + bias[n,o] )
// rows = RR (n-major), K = 3840, O in {128 (gate,sigmoid) , 64 (cand,tanh)}.
// Macro-tile 128x64, BK=32, 8 waves (4x2), wave = 32x32 (2x2 WMMA tiles).
// Every 32-wide k-chunk lies inside a single (d, hop) segment.
// ---------------------------------------------------------------------------
__global__ __launch_bounds__(256) void pool_gemm_kernel(const __bf16* __restrict__ xs,
                                                        const __bf16* __restrict__ axs,
                                                        const __bf16* __restrict__ txs,
                                                        const __bf16* __restrict__ pool, // [KTOT,O]
                                                        const float* __restrict__ E,     // [N,DE]
                                                        const float* __restrict__ bias,  // [N,O]
                                                        float* __restrict__ out,         // [RR,O]
                                                        int O, int act /*0:sigmoid 1:tanh*/) {
  __shared__ alignas(16) __bf16 lA[128 * 32];   // row-major
  __shared__ alignas(16) __bf16 lB[64 * 32];    // col-major
  const int tid = threadIdx.x;
  const int lane = tid & 31;
  const int wave = tid >> 5;
  const int wr = wave >> 1;          // 0..3
  const int wc = wave & 1;           // 0..1
  const int rowBase = blockIdx.y * 128;
  const int colBase = blockIdx.x * 64;
  const __bf16* segp[3] = {xs, axs, txs};

  v8f acc[2][2];
#pragma unroll
  for (int i = 0; i < 2; ++i)
#pragma unroll
    for (int j = 0; j < 2; ++j) acc[i][j] = vzero8();

  for (int k0 = 0; k0 < KTOT; k0 += 32) {
    const int d = k0 / (3 * CPAD);
    const int rem = k0 - d * (3 * CPAD);
    const int seg = rem >> 7;
    const int cbase = rem & (CPAD - 1);
    const __bf16* S = segp[seg];
    {   // stage activation tile 128x32 with per-row E[n,d] scaling
      const int idx = tid * 16;
      const int r = idx >> 5;
      const int kk = idx & 31;
      const int grow = rowBase + r;
      const float scale = E[(grow >> 5) * DEE + d];
      const __bf16* g = S + (size_t)grow * CPAD + cbase + kk;
      alignas(16) __bf16 tmp[16];
      *(uint4*)&tmp[0] = *(const uint4*)(g);
      *(uint4*)&tmp[8] = *(const uint4*)(g + 8);
#pragma unroll
      for (int i = 0; i < 16; ++i) tmp[i] = (__bf16)((float)tmp[i] * scale);
      *(uint4*)&lA[r * 32 + kk] = *(uint4*)&tmp[0];
      *(uint4*)&lA[r * 32 + kk + 8] = *(uint4*)&tmp[8];
    }
    {   // stage pool tile 32x64, transpose: 2 K-rows packed per ds_store_b32
      const int r0 = (tid >> 4) * 2;      // 0,2,...,30
      const int c0 = (tid & 15) * 4;      // 0..60
      const __bf16* g0 = pool + (size_t)(k0 + r0) * O + colBase + c0;
      alignas(8) unsigned short ta[4], tb[4];
      *(uint2*)ta = *(const uint2*)(g0);
      *(uint2*)tb = *(const uint2*)(g0 + O);
      if (k0 + 32 < KTOT) __builtin_prefetch(g0 + (size_t)32 * O, 0, 1);
#pragma unroll
      for (int i = 0; i < 4; ++i) {
        unsigned pv = (unsigned)ta[i] | ((unsigned)tb[i] << 16);
        *(unsigned*)&lB[(c0 + i) * 32 + r0] = pv;
      }
    }
    __syncthreads();
    v16bf bf[2];
#pragma unroll
    for (int tn = 0; tn < 2; ++tn)
      bf[tn] = load_b_frag(&lB[(wc * 32 + tn * 16) * 32], lane);
#pragma unroll
    for (int tm = 0; tm < 2; ++tm) {
      v16bf a = load_a_frag(&lA[(wr * 32 + tm * 16) * 32], lane);
#pragma unroll
      for (int tn = 0; tn < 2; ++tn)
        acc[tm][tn] = __builtin_amdgcn_wmma_f32_16x16x32_bf16(
            false, a, false, bf[tn], (short)0, acc[tm][tn], false, false);
    }
    __syncthreads();
  }

  const int hi = (lane >> 4) & 1;
  const int cl = lane & 15;
#pragma unroll
  for (int tm = 0; tm < 2; ++tm)
#pragma unroll
    for (int tn = 0; tn < 2; ++tn) {
      const int col = colBase + wc * 32 + tn * 16 + cl;
#pragma unroll
      for (int j = 0; j < 8; ++j) {
        const int row = rowBase + wr * 32 + tm * 16 + j + hi * 8;
        const int n = row >> 5;
        float v = acc[tm][tn][j] + bias[(size_t)n * O + col];
        v = act ? tanhf(v) : (1.f / (1.f + __expf(-v)));
        out[(size_t)row * O + col] = v;
      }
    }
}

// ---------------------------------------------------------------------------
// K8: GRU state update  h = r*h + (1-r)*hc ; optionally emit output sequence
// ---------------------------------------------------------------------------
__global__ void gru_update_kernel(float* __restrict__ state, const float* __restrict__ zr,
                                  const float* __restrict__ hc, float* __restrict__ outSeq,
                                  int t) {
  int idx = blockIdx.x * blockDim.x + threadIdx.x;
  if (idx >= RR * HH) return;
  int row = idx >> 6;
  int h = idx & (HH - 1);
  float r = zr[(size_t)row * CPAD + HH + h];
  float v = r * state[idx] + (1.f - r) * hc[idx];
  state[idx] = v;
  if (outSeq) {
    int n = row >> 5, b = row & 31;
    outSeq[(((size_t)b * TT + t) * NN + n) * HH + h] = v;
  }
}

// K9: finals [L=2, B, N, H]
__global__ void copy_final_kernel(const float* __restrict__ s0, const float* __restrict__ s1,
                                  float* __restrict__ dst) {
  int idx = blockIdx.x * blockDim.x + threadIdx.x;
  if (idx >= 2 * RR * HH) return;
  int l = idx / (RR * HH);
  int rem = idx - l * (RR * HH);
  int row = rem >> 6;
  int h = rem & (HH - 1);
  int n = row >> 5, b = row & 31;
  dst[(((size_t)l * BB + b) * NN + n) * HH + h] = (l ? s1 : s0)[rem];
}

// ---------------------------------------------------------------------------
extern "C" void kernel_launch(void* const* d_in, const int* in_sizes, int n_in,
                              void* d_out, int out_size, void* d_ws, size_t ws_size,
                              hipStream_t stream) {
  (void)in_sizes; (void)n_in; (void)out_size; (void)ws_size;
  const float* x     = (const float*)d_in[0];   // [B,T,N,1]
  const float* E     = (const float*)d_in[1];   // [N,DE]
  const float* tilde = (const float*)d_in[2];   // [N,N]
  const float* gw[2] = {(const float*)d_in[3], (const float*)d_in[7]};
  const float* gb[2] = {(const float*)d_in[4], (const float*)d_in[8]};
  const float* uw[2] = {(const float*)d_in[5], (const float*)d_in[9]};
  const float* ub[2] = {(const float*)d_in[6], (const float*)d_in[10]};
  float* out = (float*)d_out;

  char* ws = (char*)d_ws;
  size_t off = 0;
  auto alloc = [&](size_t bytes) -> char* {
    char* p = ws + off;
    off += (bytes + 255) & ~(size_t)255;
    return p;
  };
  __bf16* A_bf = (__bf16*)alloc((size_t)NN * NN * 2);
  __bf16* T_bf = (__bf16*)alloc((size_t)NN * NN * 2);
  __bf16* poolG[2], *poolU[2];
  float *bG[2], *bU[2];
  for (int l = 0; l < 2; ++l) {
    poolG[l] = (__bf16*)alloc((size_t)KTOT * 128 * 2);
    poolU[l] = (__bf16*)alloc((size_t)KTOT * 64 * 2);
  }
  for (int l = 0; l < 2; ++l) {
    bG[l] = (float*)alloc((size_t)NN * 128 * 4);
    bU[l] = (float*)alloc((size_t)NN * 64 * 4);
  }
  __bf16* xs  = (__bf16*)alloc((size_t)RR * CPAD * 2);
  __bf16* axs = (__bf16*)alloc((size_t)RR * CPAD * 2);
  __bf16* txs = (__bf16*)alloc((size_t)RR * CPAD * 2);
  float* zr   = (float*)alloc((size_t)RR * CPAD * 4);
  float* hc   = (float*)alloc((size_t)RR * HH * 4);
  float* st[2];
  st[0] = (float*)alloc((size_t)RR * HH * 4);
  st[1] = (float*)alloc((size_t)RR * HH * 4);

  const int TPB = 256;
  // ---- one-time prep (recomputed every call for determinism) ----
  zero_f32_kernel<<<(RR * HH + TPB - 1) / TPB, TPB, 0, stream>>>(st[0], RR * HH);
  zero_f32_kernel<<<(RR * HH + TPB - 1) / TPB, TPB, 0, stream>>>(st[1], RR * HH);
  adj_softmax_kernel<<<NN, TPB, 0, stream>>>(E, A_bf);
  cvt_bf16_kernel<<<(NN * NN + TPB - 1) / TPB, TPB, 0, stream>>>(tilde, T_bf, NN * NN);
  const int cin[2] = {HH + 1, 2 * HH};
  for (int l = 0; l < 2; ++l) {
    cvt_pool_kernel<<<(KTOT * 128 + TPB - 1) / TPB, TPB, 0, stream>>>(gw[l], poolG[l], cin[l], 128);
    cvt_pool_kernel<<<(KTOT * 64 + TPB - 1) / TPB, TPB, 0, stream>>>(uw[l], poolU[l], cin[l], 64);
    bias_kernel<<<(NN * 128 + TPB - 1) / TPB, TPB, 0, stream>>>(E, gb[l], bG[l], 128);
    bias_kernel<<<(NN * 64 + TPB - 1) / TPB, TPB, 0, stream>>>(E, ub[l], bU[l], 64);
  }

  const dim3 propGrid(WCOLS / 128, NN / 128);        // 32 x 16
  const dim3 gateGrid(2, RR / 128);                  // O=128
  const dim3 candGrid(1, RR / 128);                  // O=64
  const int buildBlocks = (RR * CPAD + TPB - 1) / TPB;
  const int updBlocks = (RR * HH + TPB - 1) / TPB;

  // ---- 12-step scan, layers interleaved per step ----
  for (int t = 0; t < TT; ++t) {
    for (int l = 0; l < 2; ++l) {
      const float* xsrc = (l == 0) ? x : nullptr;
      const float* hprev = (l == 0) ? nullptr : st[0];
      // gate path: xs = [x_t | h] ; z_r = sigmoid(gconv(xs))
      build_in_kernel<<<buildBlocks, TPB, 0, stream>>>(xsrc, st[l], hprev, nullptr, xs, l, 0, t);
      prop_gemm_kernel<<<propGrid, TPB, 0, stream>>>(A_bf, xs, axs);
      prop_gemm_kernel<<<propGrid, TPB, 0, stream>>>(T_bf, xs, txs);
      pool_gemm_kernel<<<gateGrid, TPB, 0, stream>>>(xs, axs, txs, poolG[l], E, bG[l], zr, 128, 0);
      // candidate path: xs = [x_t | z*h] ; hc = tanh(gconv(xs))
      build_in_kernel<<<buildBlocks, TPB, 0, stream>>>(xsrc, st[l], hprev, zr, xs, l, 1, t);
      prop_gemm_kernel<<<propGrid, TPB, 0, stream>>>(A_bf, xs, axs);
      prop_gemm_kernel<<<propGrid, TPB, 0, stream>>>(T_bf, xs, txs);
      pool_gemm_kernel<<<candGrid, TPB, 0, stream>>>(xs, axs, txs, poolU[l], E, bU[l], hc, 64, 1);
      // h = r*h + (1-r)*hc ; layer1 also writes the output sequence
      gru_update_kernel<<<updBlocks, TPB, 0, stream>>>(st[l], zr, hc,
                                                       (l == 1) ? out : nullptr, t);
    }
  }
  copy_final_kernel<<<(2 * RR * HH + TPB - 1) / TPB, TPB, 0, stream>>>(
      st[0], st[1], out + (size_t)BB * TT * NN * HH);
}
